// LearnablePixelwiseAnisoJBU_NoParent_70437463654777
// MI455X (gfx1250) — matrix operand order repaired and checked by reading, not compiled, not run on hardware
//
#include <hip/hip_runtime.h>
#include <hip/hip_fp16.h>
#include <math.h>

typedef __attribute__((ext_vector_type(16))) _Float16 v16h;
typedef __attribute__((ext_vector_type(8)))  _Float16 v8h;   // 16 B: one ds_load_b128
typedef __attribute__((ext_vector_type(8)))  float    v8f;

#define HL    32
#define WL    32
#define CH    128
#define HH    512
#define WH    512
#define NK    81       // 9x9 taps
#define KPAD  96       // 3 * 32 (K padded for 16x16x32 WMMA)
#define KROW  104      // row stride in halves: 208 B, 16B-aligned, 52-dword bank stride
#define NPIX  256      // 16x16 hi-res pixels per tile
#define PLANE (HH*WH)

__global__ __launch_bounds__(256)
void jbu_wmma_kernel(const float* __restrict__ feat,
                     const float* __restrict__ guide,
                     const float* __restrict__ sx_raw,
                     const float* __restrict__ sy_raw,
                     const float* __restrict__ th_raw,
                     const float* __restrict__ sr_raw,
                     float* __restrict__ out)
{
    const int tile = blockIdx.x;      // 0..1023
    const int u = tile >> 5;          // low-res row
    const int v = tile & 31;          // low-res col
    const int tid = threadIdx.x;      // 0..255

    // W stored transposed, fragment-ready: Wt8[pixel] = 13 x v8h = 104 halves.
    __shared__ v8h   Wt8[NPIX][KROW / 8];   // 53.2 KB
    __shared__ float gl[NK * 3];            // guide_lr at 81 clipped neighbors
    __shared__ int   posk[KPAD];            // low-res linear offsets per k (pad -> 0)
    __shared__ float invden_s[NPIX];        // per-pixel 1/max(den,1e-8)

    // ---------------- Phase 1: neighbor position table + on-the-fly guide_lr ----
    // Bilinear 512->32 downsample degenerates to a 2x2 average at rows/cols
    // {16i+7,16i+8} because src = 16i + 7.5 exactly (t = 0.5).
    if (tid < KPAD) {
        int pos = 0;
        if (tid < NK) {
            int dy = tid / 9 - 4, dx = tid % 9 - 4;
            int ui = min(max(u + dy, 0), HL - 1);
            int vi = min(max(v + dx, 0), WL - 1);
            pos = ui * WL + vi;
        }
        posk[tid] = pos;
    }
    if (tid < NK * 3) {
        int c = tid / NK;
        int k = tid - c * NK;
        int dy = k / 9 - 4, dx = k % 9 - 4;
        int ui = min(max(u + dy, 0), HL - 1);
        int vi = min(max(v + dx, 0), WL - 1);
        const float* g = guide + c * PLANE;
        int r0 = ui * 16 + 7, c0 = vi * 16 + 7;
        float val = 0.25f * (g[r0 * WH + c0]       + g[r0 * WH + c0 + 1] +
                             g[(r0 + 1) * WH + c0] + g[(r0 + 1) * WH + c0 + 1]);
        gl[k * 3 + c] = val;
    }
    __syncthreads();

    // ---------------- Phase 2: per-pixel weight generation into LDS ------------
    {
        const int py = tid >> 4, px = tid & 15;
        const float Yf = (float)(u * 16 + py);
        const float Xf = (float)(v * 16 + px);

        const int pidx = u * WL + v;
        float sx = fmaxf(__expf(sx_raw[pidx]), 1e-6f);
        float sy = fmaxf(__expf(sy_raw[pidx]), 1e-6f);
        float sr = fmaxf(__expf(sr_raw[pidx]), 1e-6f);
        float th = (float)M_PI * tanhf(th_raw[pidx]);
        float ct = __cosf(th), st = __sinf(th);
        float inv2sx2 = __builtin_amdgcn_rcpf(2.0f * sx * sx + 1e-8f);
        float inv2sy2 = __builtin_amdgcn_rcpf(2.0f * sy * sy + 1e-8f);
        float inv2sr2 = __builtin_amdgcn_rcpf(2.0f * sr * sr + 1e-8f);
        float R = fminf(fmaxf(2.0f * fmaxf(sx, sy), 1.0f), 4.0f);
        float Rsq = R * R;

        const int gofs = (u * 16 + py) * WH + (v * 16 + px);
        float g0 = guide[gofs];
        float g1 = guide[PLANE + gofs];
        float g2 = guide[2 * PLANE + gofs];

        _Float16* wrow = (_Float16*)&Wt8[tid][0];
        float den = 0.0f;
        int k = 0;
        for (int dyi = 0; dyi < 9; ++dyi) {
            int dy = dyi - 4;
            int ui = min(max(u + dy, 0), HL - 1);
            float cdy = (Yf - (float)(ui * 16) - 7.5f) * 0.0625f;
            for (int dxi = 0; dxi < 9; ++dxi, ++k) {
                int dx = dxi - 4;
                int vi = min(max(v + dx, 0), WL - 1);
                float cdx = (Xf - (float)(vi * 16) - 7.5f) * 0.0625f;
                float a =  cdx * ct + cdy * st;
                float b = -cdx * st + cdy * ct;
                float d0 = g0 - gl[k * 3 + 0];
                float d1 = g1 - gl[k * 3 + 1];
                float d2 = g2 - gl[k * 3 + 2];
                float gd2 = d0 * d0 + d1 * d1 + d2 * d2;
                float logw = -(a * a) * inv2sx2 - (b * b) * inv2sy2 - gd2 * inv2sr2;
                float w = __expf(logw);
                w = ((float)(dy * dy + dx * dx) <= Rsq) ? w : 0.0f;
                den += w;
                wrow[k] = (_Float16)w;
            }
        }
        for (int kp = NK; kp < KPAD; ++kp) wrow[kp] = (_Float16)0.0f;
        invden_s[tid] = __builtin_amdgcn_rcpf(fmaxf(den, 1e-8f));
    }
    __syncthreads();

    // ---------------- Phase 3: tile GEMM, C(128x256) = F(128x96) x W(96x256) ---
    const int lane = tid & 31;
    const int wave = tid >> 5;               // M-tile index (0..7)
    const int lan15 = lane & 15;
    const int mrow = (wave << 4) + lan15;    // global channel for A fragment
    const int alane8 = (lane < 16) ? 0 : 16; // K offset for upper half-wave (A)
    const int bhalf  = (lane < 16) ? 0 : 2;  // v8h index offset (K+16) for upper half

    // Gather A fragments once per wave (reused across all 16 N-tiles).
    // A layout (16-bit 16x32): lane halves h -> K = h (+16 for lanes 16-31), contiguous like B.
    v16h A0, A1, A2;
    const float* frow = feat + mrow * (HL * WL);
#pragma unroll
    for (int h = 0; h < 16; ++h) {
        int kk = h + alane8;
        A0[h] = (_Float16)frow[posk[kk]];
        A1[h] = (_Float16)frow[posk[32 + kk]];
        A2[h] = (_Float16)frow[posk[64 + kk]];
    }

#pragma unroll 2
    for (int nt = 0; nt < 16; ++nt) {
        const int n = (nt << 4) + lan15;     // pixel index p = py*16+px; nt == py
        // B fragment: lane = N, halves = contiguous K -> two ds_load_b128 each.
        v8h lo0 = Wt8[n][bhalf + 0], hi0 = Wt8[n][bhalf + 1];
        v8h lo1 = Wt8[n][bhalf + 4], hi1 = Wt8[n][bhalf + 5];
        v8h lo2 = Wt8[n][bhalf + 8], hi2 = Wt8[n][bhalf + 9];
        v16h B0 = __builtin_shufflevector(lo0, hi0, 0,1,2,3,4,5,6,7,8,9,10,11,12,13,14,15);
        v16h B1 = __builtin_shufflevector(lo1, hi1, 0,1,2,3,4,5,6,7,8,9,10,11,12,13,14,15);
        v16h B2 = __builtin_shufflevector(lo2, hi2, 0,1,2,3,4,5,6,7,8,9,10,11,12,13,14,15);

        v8f c = {};
        c = __builtin_amdgcn_wmma_f32_16x16x32_f16(false, A0, false, B0,
                                                   (short)0, c, false, false);
        c = __builtin_amdgcn_wmma_f32_16x16x32_f16(false, A1, false, B1,
                                                   (short)0, c, false, false);
        c = __builtin_amdgcn_wmma_f32_16x16x32_f16(false, A2, false, B2,
                                                   (short)0, c, false, false);

        // Fused normalization + coalesced channel-major store.
        float inv = invden_s[n];
        const int y = (u << 4) + nt;
        const int x = (v << 4) + lan15;
        const int mofs = (lane < 16) ? 0 : 8;    // C f32 layout: M = r + mofs
        float* obase = out + ((wave << 4) + mofs) * PLANE + y * WH + x;
#pragma unroll
        for (int r = 0; r < 8; ++r) {
            obase[r * PLANE] = c[r] * inv;
        }
    }
}

extern "C" void kernel_launch(void* const* d_in, const int* in_sizes, int n_in,
                              void* d_out, int out_size, void* d_ws, size_t ws_size,
                              hipStream_t stream) {
    const float* feat   = (const float*)d_in[0]; // (1,128,32,32)
    const float* guide  = (const float*)d_in[1]; // (1,3,512,512)
    const float* sx_raw = (const float*)d_in[2]; // (1,1,32,32)
    const float* sy_raw = (const float*)d_in[3];
    const float* th_raw = (const float*)d_in[4];
    const float* sr_raw = (const float*)d_in[5];
    float* out = (float*)d_out;                  // (1,128,512,512)

    (void)in_sizes; (void)n_in; (void)out_size; (void)d_ws; (void)ws_size;

    dim3 grid(HL * WL);   // one workgroup per low-res tile (1024)
    dim3 block(256);      // 8 waves: one 16-channel M-tile per wave
    jbu_wmma_kernel<<<grid, block, 0, stream>>>(feat, guide, sx_raw, sy_raw,
                                                th_raw, sr_raw, out);
}
